// GNNTower_3384434229647
// MI455X (gfx1250) — matrix operand order repaired and checked by reading, not compiled, not run on hardware
//
#include <hip/hip_runtime.h>
#include <hip/hip_bf16.h>

typedef __attribute__((ext_vector_type(16))) _Float16 v16h;
typedef __attribute__((ext_vector_type(8)))  float    v8f;
typedef __attribute__((ext_vector_type(4)))  int      v4i;

#define AS1 __attribute__((address_space(1)))
#define AS3 __attribute__((address_space(3)))

#define N_NODES 50000
#define N_EDGES 800000
#define DIM     128
#define K2      256             // fused K: [x | agg]
#define NLAYERS 3
#define NGRAPHS 512
#define MTILES  (N_NODES / 16)  // 3125 exactly

// ---------------------------------------------------------------- utilities
__global__ __launch_bounds__(256) void zero_f4(float4* __restrict__ p, int n4) {
    int i = blockIdx.x * blockDim.x + threadIdx.x;
    int stride = gridDim.x * blockDim.x;
    float4 z = make_float4(0.f, 0.f, 0.f, 0.f);
    for (; i < n4; i += stride) p[i] = z;
}

__global__ __launch_bounds__(256) void zero_u32(unsigned* __restrict__ p, int n) {
    int i = blockIdx.x * blockDim.x + threadIdx.x;
    if (i < n) p[i] = 0u;
}

// ------------------------------------------------------- degree / counts
__global__ __launch_bounds__(256) void degree_kernel(const int* __restrict__ dst,
                                                     int* __restrict__ degi) {
    int e = blockIdx.x * blockDim.x + threadIdx.x;
    if (e < N_EDGES) atomicAdd(&degi[dst[e]], 1);
}

__global__ __launch_bounds__(256) void inv_deg_kernel(const int* __restrict__ degi,
                                                      float* __restrict__ invd) {
    int i = blockIdx.x * blockDim.x + threadIdx.x;
    if (i < N_NODES) {
        int d = degi[i];
        invd[i] = 1.0f / (float)(d > 0 ? d : 1);
    }
}

__global__ __launch_bounds__(256) void batch_count_kernel(const int* __restrict__ bvec,
                                                          int* __restrict__ cnt) {
    int i = blockIdx.x * blockDim.x + threadIdx.x;
    if (i < N_NODES) atomicAdd(&cnt[bvec[i]], 1);
}

// ------------------------------------------- pack W = [Ws | Wn] into f16, bc = bs+bn
__global__ __launch_bounds__(256) void prep_weights_kernel(
        const float* __restrict__ Ws, const float* __restrict__ bs,
        const float* __restrict__ Wn, const float* __restrict__ bn,
        _Float16* __restrict__ Wc, float* __restrict__ bc) {
    int idx = blockIdx.x * blockDim.x + threadIdx.x;   // NLAYERS*128*256
    if (idx < NLAYERS * DIM * K2) {
        int l = idx / (DIM * K2);
        int r = idx - l * (DIM * K2);
        int o = r / K2;           // output feature (row of Wc)
        int k = r - o * K2;       // fused K index
        float v = (k < DIM) ? Ws[(size_t)l * DIM * DIM + (size_t)o * DIM + k]
                            : Wn[(size_t)l * DIM * DIM + (size_t)o * DIM + (k - DIM)];
        Wc[idx] = (_Float16)v;
    }
    if (idx < NLAYERS * DIM) {
        int l = idx / DIM, o = idx - l * DIM;
        bc[idx] = bs[l * DIM + o] + bn[l * DIM + o];
    }
}

// ------------------------------------------------- edge scatter-add (agg += x[src])
// one wave32 per edge; lane handles 4 consecutive floats -> 512B coalesced per edge
__global__ __launch_bounds__(256) void scatter_add_kernel(
        const float* __restrict__ x, const int* __restrict__ src,
        const int* __restrict__ dst, float* __restrict__ agg) {
    int t = blockIdx.x * blockDim.x + threadIdx.x;
    int e = t >> 5;
    if (e >= N_EDGES) return;
    int lane = t & 31;
    int s = src[e];
    int d = dst[e];
    float4 v = ((const float4*)(x + (size_t)s * DIM))[lane];
    float* ap = agg + (size_t)d * DIM + lane * 4;
    atomicAdd(ap + 0, v.x);
    atomicAdd(ap + 1, v.y);
    atomicAdd(ap + 2, v.z);
    atomicAdd(ap + 3, v.w);
}

// ------------------------------------------------- fused dual-GEMM + bias + ReLU
// x_out[m,:] = relu( [x[m,:] | agg[m,:]*invdeg[m]] @ Wc^T + bc )
// In-place safe: row m's output depends only on row m's input, and the owning
// wave reads all K before its epilogue writes.
// Weights staged once per block into LDS (async global->LDS path when exposed);
// A fragments preloaded into registers; 64 WMMAs stream B from LDS only.
__global__ __launch_bounds__(256) void gemm_relu_kernel(
        const float* __restrict__ xin, const float* __restrict__ agg,
        const float* __restrict__ invd, const _Float16* __restrict__ W,
        const float* __restrict__ bias, float* __restrict__ xout) {
    __shared__ _Float16 sW[DIM * K2];       // 64 KB: whole layer weight matrix

    // ---- stage weights into LDS: 256 threads x 16 iters x 16B = 64 KB
    {
        const int t = threadIdx.x;
#if __has_builtin(__builtin_amdgcn_global_load_async_to_lds_b128) && \
    __has_builtin(__builtin_amdgcn_s_wait_asynccnt)
        #pragma unroll
        for (int i = 0; i < 16; ++i) {
            const size_t boff = ((size_t)i * 256 + t) * 16;
            __builtin_amdgcn_global_load_async_to_lds_b128(
                (v4i*)((const char*)W + boff),
                (AS3 v4i*)((char*)sW + boff), 0, 0);
        }
        __builtin_amdgcn_s_wait_asynccnt(0);
#else
        const float4* gsrc = (const float4*)W;
        float4* ldst = (float4*)sW;
        #pragma unroll
        for (int i = 0; i < 16; ++i)
            ldst[i * 256 + t] = gsrc[i * 256 + t];
#endif
    }
    __syncthreads();

    const int lane  = threadIdx.x & 31;
    const int wv    = threadIdx.x >> 5;
    const int mtile = blockIdx.x * 8 + wv;
    if (mtile < MTILES) {                   // wave-uniform: EXEC all-ones inside
        const int hi  = lane >> 4;          // half-wave select
        const int l16 = lane & 15;
        const int row = mtile * 16 + l16;   // A-matrix row owned by this lane

        const float id = invd[row];

        // ---- preload all 8 A fragments (the lane's fused 256-K row), f32->f16
        // A k order used here (same as B): halves 0..7 -> K = base+(hi?16:0)+j,
        //                        halves 8..15 -> K = base+8+(hi?16:0)+j
        v16h afrag[8];
        #pragma unroll
        for (int ks = 0; ks < 8; ++ks) {
            const bool fromAgg = (ks >= 4);
            const float* srcp = fromAgg ? (agg + (size_t)row * DIM + (ks - 4) * 32)
                                        : (xin + (size_t)row * DIM + ks * 32);
            const float scale = fromAgg ? id : 1.0f;
            v8f g0 = *(const v8f*)(srcp + (hi ? 16 : 0));
            v8f g1 = *(const v8f*)(srcp + 8 + (hi ? 16 : 0));
            #pragma unroll
            for (int j = 0; j < 8; ++j) {
                afrag[ks][j]     = (_Float16)(g0[j] * scale);
                afrag[ks][8 + j] = (_Float16)(g1[j] * scale);
            }
        }

        // ---- 64 WMMAs; B fragments stream from LDS (fused-K base = ks*32)
        v8f c[8] = {};
        #pragma unroll
        for (int n = 0; n < 8; ++n) {
            // B k order used here (same as A): lane col = n*16+l16; lanes 0-15 hold
            // K kk..kk+15, lanes 16-31 hold K kk+16..kk+31 (contiguous halves)
            const _Float16* wrow = sW + (size_t)(n * 16 + l16) * K2 + hi * 16;
            #pragma unroll
            for (int ks = 0; ks < 8; ++ks) {
                v16h b = *(const v16h*)(wrow + ks * 32);
                c[n] = __builtin_amdgcn_wmma_f32_16x16x32_f16(
                           false, afrag[ks], false, b, (short)0, c[n], false, false);
            }
        }

        // ---- epilogue: C/D layout -> lane col = n*16+l16; VGPR r -> row +r+hi*8
        #pragma unroll
        for (int n = 0; n < 8; ++n) {
            const int col = n * 16 + l16;
            const float bcol = bias[col];
            #pragma unroll
            for (int r = 0; r < 8; ++r) {
                const int orow = mtile * 16 + r + hi * 8;
                float v = c[n][r] + bcol;
                xout[(size_t)orow * DIM + col] = v > 0.0f ? v : 0.0f;
            }
        }
    }
}

// ------------------------------------------------- graph pooling
__global__ __launch_bounds__(256) void pool_scatter_kernel(
        const float* __restrict__ x, const int* __restrict__ bvec,
        float* __restrict__ pool) {
    int t = blockIdx.x * blockDim.x + threadIdx.x;
    int node = t >> 5;
    if (node >= N_NODES) return;
    int lane = t & 31;
    int g = bvec[node];
    float4 v = ((const float4*)(x + (size_t)node * DIM))[lane];
    float* pp = pool + (size_t)g * DIM + lane * 4;
    atomicAdd(pp + 0, v.x);
    atomicAdd(pp + 1, v.y);
    atomicAdd(pp + 2, v.z);
    atomicAdd(pp + 3, v.w);
}

__global__ __launch_bounds__(256) void pool_final_kernel(
        const float* __restrict__ pool, const int* __restrict__ cnt,
        float* __restrict__ out) {
    int i = blockIdx.x * blockDim.x + threadIdx.x;
    if (i < NGRAPHS * DIM) {
        int g = i / DIM;
        int cn = cnt[g];
        out[i] = pool[i] / (float)(cn > 0 ? cn : 1);
    }
}

// ================================================================ launcher
extern "C" void kernel_launch(void* const* d_in, const int* in_sizes, int n_in,
                              void* d_out, int out_size, void* d_ws, size_t ws_size,
                              hipStream_t stream) {
    const float* x0   = (const float*)d_in[0];
    const int*   ei   = (const int*)d_in[1];      // (2, E): src row 0, dst row 1
    const int*   src  = ei;
    const int*   dst  = ei + N_EDGES;
    const int*   bvec = (const int*)d_in[2];
    const float* Ws   = (const float*)d_in[4];
    const float* bs   = (const float*)d_in[5];
    const float* Wn   = (const float*)d_in[6];
    const float* bn   = (const float*)d_in[7];

    // ---- workspace carve-out (256B aligned slices)
    auto align256 = [](size_t v) { return (v + 255) & ~(size_t)255; };
    char*  base = (char*)d_ws;
    size_t off  = 0;
    float* xA   = (float*)(base + off); off = align256(off + (size_t)N_NODES * DIM * 4);
    float* agg  = (float*)(base + off); off = align256(off + (size_t)N_NODES * DIM * 4);
    float* invd = (float*)(base + off); off = align256(off + (size_t)N_NODES * 4);
    int*   degi = (int*)  (base + off); off = align256(off + (size_t)N_NODES * 4);
    int*   cnt  = (int*)  (base + off); off = align256(off + (size_t)NGRAPHS * 4);
    float* pool = (float*)(base + off); off = align256(off + (size_t)NGRAPHS * DIM * 4);
    float* bc   = (float*)(base + off); off = align256(off + (size_t)NLAYERS * DIM * 4);
    _Float16* Wc = (_Float16*)(base + off); off = align256(off + (size_t)NLAYERS * DIM * K2 * 2);

    float* outp = (float*)d_out;

    const int B = 256;

    // ---- per-call init (deterministic; ws is not re-zeroed by harness)
    zero_u32<<<(N_NODES + B - 1) / B, B, 0, stream>>>((unsigned*)degi, N_NODES);
    zero_u32<<<(NGRAPHS + B - 1) / B, B, 0, stream>>>((unsigned*)cnt, NGRAPHS);
    zero_f4<<<256, B, 0, stream>>>((float4*)pool, NGRAPHS * DIM / 4);

    degree_kernel<<<N_EDGES / B, B, 0, stream>>>(dst, degi);
    inv_deg_kernel<<<(N_NODES + B - 1) / B, B, 0, stream>>>(degi, invd);
    batch_count_kernel<<<(N_NODES + B - 1) / B, B, 0, stream>>>(bvec, cnt);
    prep_weights_kernel<<<(NLAYERS * DIM * K2 + B - 1) / B, B, 0, stream>>>(
        Ws, bs, Wn, bn, Wc, bc);

    // ---- layers (layer 0 reads const input, then in-place on xA)
    for (int l = 0; l < NLAYERS; ++l) {
        const float* xin = (l == 0) ? x0 : xA;
        zero_f4<<<2048, B, 0, stream>>>((float4*)agg, N_NODES * DIM / 4);
        scatter_add_kernel<<<(N_EDGES * 32) / B, B, 0, stream>>>(xin, src, dst, agg);
        gemm_relu_kernel<<<(MTILES + 7) / 8, B, 0, stream>>>(
            xin, agg, invd, Wc + (size_t)l * DIM * K2, bc + (size_t)l * DIM, xA);
    }

    // ---- pooling
    pool_scatter_kernel<<<(N_NODES * 32) / B, B, 0, stream>>>(xA, bvec, pool);
    pool_final_kernel<<<(NGRAPHS * DIM) / B, B, 0, stream>>>(pool, cnt, outp);
}
